// LCNetBlock_39788577030343
// MI455X (gfx1250) — matrix operand order repaired and checked by reading, not compiled, not run on hardware
//
#include <hip/hip_runtime.h>
#include <math.h>

typedef __attribute__((ext_vector_type(16))) _Float16 v16h;
typedef __attribute__((ext_vector_type(8)))  _Float16 v8h;
typedef __attribute__((ext_vector_type(8)))  float    v8f;

#define BN_RSQRT 0.99999500003749967f   /* rsqrt(1 + 1e-5) */
#define ATTN_SCALE 0.125f               /* 1/sqrt(64) */

__device__ __forceinline__ v16h zero16() {
    v16h z;
#pragma unroll
    for (int j = 0; j < 16; ++j) z[j] = (_Float16)0.f;
    return z;
}
__device__ __forceinline__ v8f zero8() {
    v8f z;
#pragma unroll
    for (int j = 0; j < 8; ++j) z[j] = 0.f;
    return z;
}

// A-fragment for V_WMMA_F32_16X16X32_F16 (A is 16x32, row-major source, K contiguous).
// lane<16 : row=lane,   holds K = 0..7 (v0..v3) and 16..23 (v4..v7)
// lane>=16: row=lane-16, holds K = 8..15 and 24..31
__device__ __forceinline__ v16h load_a_frag(const _Float16* base /* A + row*lda + k0 */, int g) {
    v8h lo = *(const v8h*)(base + 8 * g);
    v8h hi = *(const v8h*)(base + 16 + 8 * g);
    v16h r;
#pragma unroll
    for (int j = 0; j < 8; ++j) { r[j] = lo[j]; r[8 + j] = hi[j]; }
    return r;
}
// B-fragment (B is 32x16 K-major; source stored as Bt[N][K] row-major, K contiguous).
// lane<16: col=lane, K=0..7,16..23 ; lane>=16: col=lane-16, K=8..15,24..31
__device__ __forceinline__ v16h load_b_frag(const _Float16* base /* Bt + n*ldb + k0 */, int g) {
    v8h lo = *(const v8h*)(base + 8 * g);
    v8h hi = *(const v8h*)(base + 16 + 8 * g);
    v16h r;
#pragma unroll
    for (int j = 0; j < 8; ++j) { r[j] = lo[j]; r[8 + j] = hi[j]; }
    return r;
}
__device__ __forceinline__ v8f wmma_step(v16h a, v16h b, v8f c) {
    return __builtin_amdgcn_wmma_f32_16x16x32_f16(false, a, false, b, (short)0, c, false, false);
}

// ---------------------------------------------------------------------------
// Generic batched NT GEMM: C[M,N] = A[M,K] * Bt[N,K]^T (+bias) ; one wave = 32x32 tile.
// ---------------------------------------------------------------------------
__global__ void gemm_nt(const _Float16* __restrict__ A, int lda, long sA,
                        const _Float16* __restrict__ Bt, int ldb, long sB,
                        float* __restrict__ C, int ldc, long sC,
                        _Float16* __restrict__ H16, int ldh, long sH,
                        const float* __restrict__ bias, int relu, int K)
{
    const int lane = threadIdx.x;
    const int g = lane >> 4, l = lane & 15;
    const int m0 = blockIdx.x * 32, n0 = blockIdx.y * 32;
    const int z = blockIdx.z;
    A  += (long)z * sA;
    Bt += (long)z * sB;

    v8f acc[2][2];
    acc[0][0] = zero8(); acc[0][1] = zero8(); acc[1][0] = zero8(); acc[1][1] = zero8();

    for (int k0 = 0; k0 < K; k0 += 32) {
        v16h a0 = load_a_frag(A + (long)(m0 + l) * lda + k0, g);
        v16h a1 = load_a_frag(A + (long)(m0 + 16 + l) * lda + k0, g);
        v16h b0 = load_b_frag(Bt + (long)(n0 + l) * ldb + k0, g);
        v16h b1 = load_b_frag(Bt + (long)(n0 + 16 + l) * ldb + k0, g);
        acc[0][0] = wmma_step(a0, b0, acc[0][0]);
        acc[0][1] = wmma_step(a0, b1, acc[0][1]);
        acc[1][0] = wmma_step(a1, b0, acc[1][0]);
        acc[1][1] = wmma_step(a1, b1, acc[1][1]);
    }
#pragma unroll
    for (int mi = 0; mi < 2; ++mi)
#pragma unroll
        for (int ni = 0; ni < 2; ++ni) {
            const int col = n0 + 16 * ni + l;
            const float bi = bias ? bias[col] : 0.f;
#pragma unroll
            for (int i = 0; i < 8; ++i) {
                const int row = m0 + 16 * mi + 8 * g + i;
                float v = acc[mi][ni][i] + bi;
                if (relu) v = fmaxf(v, 0.f);
                if (C)   C[(long)z * sC + (long)row * ldc + col] = v;
                if (H16) H16[(long)z * sH + (long)row * ldh + col] = (_Float16)v;
            }
        }
}

// ---------------------------------------------------------------------------
// Conv as implicit GEMM over taps. Apk: activations [B*H*W][Cin] f16 (H=W pow2).
// Wpk: [KH*KW][Nc][Cin] f16. Output ldc = Nc (fp32) and/or H16 with ldh.
// ---------------------------------------------------------------------------
__global__ void conv_gemm(const _Float16* __restrict__ Apk,
                          const _Float16* __restrict__ Wpk,
                          float* __restrict__ C32,
                          _Float16* __restrict__ H16, int ldh,
                          const float* __restrict__ bias,
                          const float* __restrict__ bng, const float* __restrict__ bnb,
                          int relu, int Cin, int Nc,
                          int logHW, int logW, int KH, int KW, int pad)
{
    const int lane = threadIdx.x;
    const int g = lane >> 4, l = lane & 15;
    const int m0 = blockIdx.x * 32, n0 = blockIdx.y * 32;
    const int Wd = 1 << logW;
    const int Hd = 1 << (logHW - logW);

    int bb[2], yy[2], xx[2];
#pragma unroll
    for (int i = 0; i < 2; ++i) {
        const int m = m0 + 16 * i + l;
        bb[i] = m >> logHW;
        const int rem = m & ((1 << logHW) - 1);
        yy[i] = rem >> logW;
        xx[i] = rem & (Wd - 1);
    }

    v8f acc[2][2];
    acc[0][0] = zero8(); acc[0][1] = zero8(); acc[1][0] = zero8(); acc[1][1] = zero8();

    const int T = KH * KW;
    for (int t = 0; t < T; ++t) {
        const int dy = t / KW - pad;
        const int dx = t % KW - pad;
        const _Float16* rp[2];
#pragma unroll
        for (int i = 0; i < 2; ++i) {
            const int y2 = yy[i] + dy, x2 = xx[i] + dx;
            const bool ok = ((unsigned)y2 < (unsigned)Hd) && ((unsigned)x2 < (unsigned)Wd);
            rp[i] = ok ? (Apk + (long)((bb[i] << logHW) + (y2 << logW) + x2) * Cin) : nullptr;
        }
        const _Float16* wt = Wpk + (long)t * Nc * Cin;
        for (int k0 = 0; k0 < Cin; k0 += 32) {
            v16h a0 = rp[0] ? load_a_frag(rp[0] + k0, g) : zero16();
            v16h a1 = rp[1] ? load_a_frag(rp[1] + k0, g) : zero16();
            v16h b0 = load_b_frag(wt + (long)(n0 + l) * Cin + k0, g);
            v16h b1 = load_b_frag(wt + (long)(n0 + 16 + l) * Cin + k0, g);
            acc[0][0] = wmma_step(a0, b0, acc[0][0]);
            acc[0][1] = wmma_step(a0, b1, acc[0][1]);
            acc[1][0] = wmma_step(a1, b0, acc[1][0]);
            acc[1][1] = wmma_step(a1, b1, acc[1][1]);
        }
    }
#pragma unroll
    for (int mi = 0; mi < 2; ++mi)
#pragma unroll
        for (int ni = 0; ni < 2; ++ni) {
            const int col = n0 + 16 * ni + l;
            const float bi = bias ? bias[col] : 0.f;
            float sc = 1.f, sh = 0.f;
            if (bng) { sc = bng[col] * BN_RSQRT; sh = bnb[col]; }
#pragma unroll
            for (int i = 0; i < 8; ++i) {
                const int row = m0 + 16 * mi + 8 * g + i;
                float v = acc[mi][ni][i] + bi;
                if (bng) v = v * sc + sh;
                if (relu) v = fmaxf(v, 0.f);
                if (C32) C32[(long)row * Nc + col] = v;
                if (H16) H16[(long)row * ldh + col] = (_Float16)v;
            }
        }
}

// ---------------------------------------------------------------------------
// Packing / pointwise / reduction kernels
// ---------------------------------------------------------------------------
__global__ void pack_x_kernel(const float* __restrict__ x, _Float16* __restrict__ dst) {
    const int id = blockIdx.x * 256 + threadIdx.x;          // 16*4096*64
    const int c = id & 63;
    const int p = id >> 6;                                  // global pixel b*4096+pix
    const int b = p >> 12, pix = p & 4095;
    dst[id] = (_Float16)x[((long)(b * 64 + c) << 12) + pix];
}

__global__ void packw_kernel(const float* __restrict__ w, _Float16* __restrict__ dst,
                             int Cout, int Cin, int T, int n) {
    const int id = blockIdx.x * 256 + threadIdx.x;
    if (id >= n) return;
    const int ci = id % Cin;
    const int r = id / Cin;
    const int co = r % Cout;
    const int t = r / Cout;
    dst[id] = (_Float16)w[(long)(co * Cin + ci) * T + t];
}

__global__ void sc_kernel(const float* __restrict__ x, const float* __restrict__ w,
                          const float* __restrict__ bias, const float* __restrict__ g,
                          const float* __restrict__ bt, float* __restrict__ out) {
    const int id = blockIdx.x * 256 + threadIdx.x;          // 16384*128
    const int c = id & 127;
    const int p = id >> 7;
    const int b = p >> 10, y = (p >> 5) & 31, xx = p & 31;
    const float* xb = x + ((long)b * 64) * 4096 + (2 * y) * 64 + 2 * xx;
    float s = 0.f;
#pragma unroll 4
    for (int ci = 0; ci < 64; ++ci) s += xb[(long)ci * 4096] * w[c * 64 + ci];
    s += bias[c];
    out[id] = s * (g[c] * BN_RSQRT) + bt[c];
}

__global__ void dwconv_kernel(const float* __restrict__ in, int Hin, int Win,
                              float* __restrict__ outP, _Float16* __restrict__ outH,
                              float* __restrict__ outNCHW, const float* __restrict__ resid,
                              int Hout, int Wout,
                              const float* __restrict__ w, const float* __restrict__ bias,
                              const float* __restrict__ bng, const float* __restrict__ bnb,
                              int relu, int chanOff, int nchan, int ks, int pad, int stride,
                              int total)
{
    const int id = blockIdx.x * 256 + threadIdx.x;
    if (id >= total) return;
    const int c = id % nchan;
    int r = id / nchan;
    const int xo = r % Wout; r /= Wout;
    const int yo = r % Hout;
    const int b = r / Hout;
    const float* wc = w + c * ks * ks;
    float acc = 0.f;
    for (int ky = 0; ky < ks; ++ky) {
        const int yi = yo * stride + ky - pad;
        if ((unsigned)yi >= (unsigned)Hin) continue;
        for (int kx = 0; kx < ks; ++kx) {
            const int xi = xo * stride + kx - pad;
            if ((unsigned)xi >= (unsigned)Win) continue;
            acc += in[(((long)b * Hin + yi) * Win + xi) * 128 + chanOff + c] * wc[ky * ks + kx];
        }
    }
    float v = acc + bias[c];
    if (bng) v = v * (bng[c] * BN_RSQRT) + bnb[c];
    if (relu) v = fmaxf(v, 0.f);
    const long opix = ((long)b * Hout + yo) * Wout + xo;
    if (resid) v += resid[opix * 128 + chanOff + c];
    if (outP) outP[opix * 128 + chanOff + c] = v;
    if (outH) outH[opix * 128 + chanOff + c] = (_Float16)v;
    if (outNCHW) outNCHW[(((long)b * 128 + chanOff + c) * Hout + yo) * Wout + xo] = v;
}

__global__ void gfeat_kernel(const float* __restrict__ xm, float* __restrict__ gfeat) {
    const int b = blockIdx.x, c = threadIdx.x;               // 16 x 128
    float s = 0.f;
    for (int p = 0; p < 1024; ++p) s += xm[((long)b * 1024 + p) * 128 + c];
    gfeat[b * 128 + c] = s * (1.f / 1024.f);
}

__global__ void gate_kernel(const float* __restrict__ gfeat, const float* __restrict__ fcw,
                            const float* __restrict__ fcb, float* __restrict__ gmask) {
    const int b = threadIdx.x;
    if (b >= 16) return;
    float l[3];
    for (int j = 0; j < 3; ++j) {
        float s = fcb[j];
        for (int c = 0; c < 128; ++c) s += gfeat[b * 128 + c] * fcw[j * 128 + c];
        l[j] = s;
    }
    const float mx = fmaxf(l[0], fmaxf(l[1], l[2]));
    float e[3], sum = 0.f;
    for (int j = 0; j < 3; ++j) { e[j] = expf(l[j] - mx); sum += e[j]; }
    float p[3];
    for (int j = 0; j < 3; ++j) p[j] = e[j] / sum;
    int mn = 0;
    if (p[1] < p[mn]) mn = 1;
    if (p[2] < p[mn]) mn = 2;                                  // drop smallest = top-2 keep
    for (int j = 0; j < 3; ++j) gmask[b * 3 + j] = (j == mn) ? 0.f : p[j];
}

__global__ void softmax_rows(float* __restrict__ S, int ncol, float scale) {
    float* r = S + (long)blockIdx.x * ncol;
    __shared__ float red[256];
    const int t = threadIdx.x;
    float mx = -1e30f;
    for (int i = t; i < ncol; i += 256) mx = fmaxf(mx, r[i] * scale);
    red[t] = mx; __syncthreads();
    for (int s = 128; s > 0; s >>= 1) { if (t < s) red[t] = fmaxf(red[t], red[t + s]); __syncthreads(); }
    mx = red[0]; __syncthreads();
    float sum = 0.f;
    for (int i = t; i < ncol; i += 256) { const float e = expf(r[i] * scale - mx); r[i] = e; sum += e; }
    red[t] = sum; __syncthreads();
    for (int s = 128; s > 0; s >>= 1) { if (t < s) red[t] += red[t + s]; __syncthreads(); }
    const float inv = 1.f / red[0];
    for (int i = t; i < ncol; i += 256) r[i] *= inv;
}

__global__ void colsum_kernel(const float* __restrict__ S, float* __restrict__ imp) {
    const int b = blockIdx.x;
    const int m = blockIdx.y * 256 + threadIdx.x;
    const float* base = S + (long)b * 1024 * 1024 + m;
    float s = 0.f;
    for (int n = 0; n < 1024; ++n) s += base[(long)n * 1024];
    imp[b * 1024 + m] = s;
}

__global__ void top2_kernel(const float* __restrict__ imp, int* __restrict__ kidx) {
    const int b = threadIdx.x;
    if (b >= 16) return;
    const float* r = imp + b * 1024;
    int i0 = 0, i1 = 0; float v0 = r[0], v1 = -1e30f;
    for (int i = 1; i < 1024; ++i) {
        const float v = r[i];
        if (v > v0) { v1 = v0; i1 = i0; v0 = v; i0 = i; }
        else if (v > v1) { v1 = v; i1 = i; }
    }
    kidx[2 * b] = i0; kidx[2 * b + 1] = i1;
}

// out[n] = attn[n,k0]*V[k0] + attn[n,k1]*V[k1]  (top-2 key mask => rank-2 update)
__global__ void sparse_av(const float* __restrict__ S, const _Float16* __restrict__ Vb,
                          const int* __restrict__ kidx, _Float16* __restrict__ out) {
    const int id = blockIdx.x * 256 + threadIdx.x;          // 16384*128
    const int c = id & 127;
    const int p = id >> 7;
    const int b = p >> 10, n = p & 1023;
    const int k0 = kidx[2 * b], k1 = kidx[2 * b + 1];
    const float* row = S + ((long)b * 1024 + n) * 1024;
    const float v = row[k0] * (float)Vb[((long)b * 1024 + k0) * 128 + c]
                  + row[k1] * (float)Vb[((long)b * 1024 + k1) * 128 + c];
    out[id] = (_Float16)v;
}

__global__ void ymul_kernel(const float* __restrict__ xm, const float* __restrict__ omd,
                            const float* __restrict__ gmask, _Float16* __restrict__ out) {
    const int id = blockIdx.x * 256 + threadIdx.x;          // 16384*128
    const int c = id & 127;
    const int b = id >> 17;                                  // (id>>7)>>10
    const int grp = (c < 43) ? 0 : ((c < 86) ? 1 : 2);
    out[id] = (_Float16)(xm[id] * gmask[b * 3 + grp] * omd[id]);
}

// ---------------------------------------------------------------------------
extern "C" void kernel_launch(void* const* d_in, const int* in_sizes, int n_in,
                              void* d_out, int out_size, void* d_ws, size_t ws_size,
                              hipStream_t stream)
{
    (void)in_sizes; (void)n_in; (void)out_size; (void)ws_size;
    const float* x      = (const float*)d_in[0];
    const float* ic1_w  = (const float*)d_in[1];
    const float* ic1_b  = (const float*)d_in[2];
    const float* ic1_g  = (const float*)d_in[3];
    const float* ic1_bt = (const float*)d_in[4];
    const float* ic2_w  = (const float*)d_in[5];
    const float* ic2_b  = (const float*)d_in[6];
    const float* ic2_g  = (const float*)d_in[7];
    const float* ic2_bt = (const float*)d_in[8];
    const float* dt3_w  = (const float*)d_in[9];
    const float* dt3_b  = (const float*)d_in[10];
    const float* dt3_g  = (const float*)d_in[11];
    const float* dt3_bt = (const float*)d_in[12];
    const float* dt5_w  = (const float*)d_in[13];
    const float* dt5_b  = (const float*)d_in[14];
    const float* dt5_g  = (const float*)d_in[15];
    const float* dt5_bt = (const float*)d_in[16];
    const float* dt7_w  = (const float*)d_in[17];
    const float* dt7_b  = (const float*)d_in[18];
    const float* dt7_g  = (const float*)d_in[19];
    const float* dt7_bt = (const float*)d_in[20];
    const float* fc_w   = (const float*)d_in[21];
    const float* fc_b   = (const float*)d_in[22];
    const float* md1_w  = (const float*)d_in[23];
    const float* md1_b  = (const float*)d_in[24];
    const float* md3_w  = (const float*)d_in[25];
    const float* md3_b  = (const float*)d_in[26];
    const float* md5_w  = (const float*)d_in[27];
    const float* md5_b  = (const float*)d_in[28];
    const float* q_w    = (const float*)d_in[29];
    const float* q_b    = (const float*)d_in[30];
    const float* k_w    = (const float*)d_in[31];
    const float* k_b    = (const float*)d_in[32];
    const float* v_w    = (const float*)d_in[33];
    const float* v_b    = (const float*)d_in[34];
    const float* o_w    = (const float*)d_in[35];
    const float* o_b    = (const float*)d_in[36];
    const float* oc1_w  = (const float*)d_in[37];
    const float* oc1_b  = (const float*)d_in[38];
    const float* oc1_g  = (const float*)d_in[39];
    const float* oc1_bt = (const float*)d_in[40];
    const float* oc2_w  = (const float*)d_in[41];
    const float* oc2_b  = (const float*)d_in[42];
    const float* oc2_g  = (const float*)d_in[43];
    const float* oc2_bt = (const float*)d_in[44];
    const float* sc_w   = (const float*)d_in[45];
    const float* sc_b   = (const float*)d_in[46];
    const float* sc_g   = (const float*)d_in[47];
    const float* sc_bt  = (const float*)d_in[48];

    // Deterministic bump allocator over d_ws
    char* w8 = (char*)d_ws;
    size_t off = 0;
    auto alloc = [&](size_t bytes) -> void* {
        void* p = (void*)(w8 + off);
        off += (bytes + 255) & ~(size_t)255;
        return p;
    };
    _Float16* Ax     = (_Float16*)alloc((size_t)65536 * 64 * 2);
    _Float16* wp_ic1 = (_Float16*)alloc((size_t)128 * 64 * 2);
    _Float16* wp_md1 = (_Float16*)alloc((size_t)128 * 128 * 2);
    _Float16* wp_md3 = (_Float16*)alloc((size_t)9 * 128 * 128 * 2);
    _Float16* wp_md5 = (_Float16*)alloc((size_t)25 * 128 * 128 * 2);
    _Float16* wp_oc1 = (_Float16*)alloc((size_t)128 * 128 * 2);
    _Float16* wp_q   = (_Float16*)alloc((size_t)64 * 384 * 2);
    _Float16* wp_k   = (_Float16*)alloc((size_t)64 * 384 * 2);
    _Float16* wp_v   = (_Float16*)alloc((size_t)128 * 384 * 2);
    _Float16* wp_o   = (_Float16*)alloc((size_t)128 * 128 * 2);
    float*    resid  = (float*)alloc((size_t)16384 * 128 * 4);
    float*    h1     = (float*)alloc((size_t)65536 * 128 * 4);
    float*    h      = (float*)alloc((size_t)16384 * 128 * 4);
    _Float16* hb     = (_Float16*)alloc((size_t)16384 * 128 * 2);
    float*    xmulti = (float*)alloc((size_t)16384 * 128 * 4);
    float*    gfeat  = (float*)alloc((size_t)16 * 128 * 4);
    float*    gmask  = (float*)alloc((size_t)16 * 3 * 4);
    _Float16* fm     = (_Float16*)alloc((size_t)16384 * 384 * 2);
    _Float16* Qb     = (_Float16*)alloc((size_t)16384 * 64 * 2);
    _Float16* Kb     = (_Float16*)alloc((size_t)16384 * 64 * 2);
    _Float16* Vb     = (_Float16*)alloc((size_t)16384 * 128 * 2);
    float*    scores = (float*)alloc((size_t)16 * 1024 * 1024 * 4);
    float*    keyimp = (float*)alloc((size_t)16 * 1024 * 4);
    int*      kidx   = (int*)alloc((size_t)16 * 2 * 4);
    _Float16* attnout= (_Float16*)alloc((size_t)16384 * 128 * 2);
    float*    out_md = (float*)alloc((size_t)16384 * 128 * 4);
    _Float16* ymul   = (_Float16*)alloc((size_t)16384 * 128 * 2);
    float*    y1     = (float*)alloc((size_t)16384 * 128 * 4);

    const dim3 B256(256), B32(32);

    // 1) pack input & weights to f16
    pack_x_kernel<<<4194304 / 256, B256, 0, stream>>>(x, Ax);
    packw_kernel<<<(8192 + 255) / 256, B256, 0, stream>>>(ic1_w, wp_ic1, 128, 64, 1, 8192);
    packw_kernel<<<(16384 + 255) / 256, B256, 0, stream>>>(md1_w, wp_md1, 128, 128, 1, 16384);
    packw_kernel<<<(147456 + 255) / 256, B256, 0, stream>>>(md3_w, wp_md3, 128, 128, 9, 147456);
    packw_kernel<<<(409600 + 255) / 256, B256, 0, stream>>>(md5_w, wp_md5, 128, 128, 25, 409600);
    packw_kernel<<<(16384 + 255) / 256, B256, 0, stream>>>(oc1_w, wp_oc1, 128, 128, 1, 16384);
    packw_kernel<<<(24576 + 255) / 256, B256, 0, stream>>>(q_w, wp_q, 64, 384, 1, 24576);
    packw_kernel<<<(24576 + 255) / 256, B256, 0, stream>>>(k_w, wp_k, 64, 384, 1, 24576);
    packw_kernel<<<(49152 + 255) / 256, B256, 0, stream>>>(v_w, wp_v, 128, 384, 1, 49152);
    packw_kernel<<<(16384 + 255) / 256, B256, 0, stream>>>(o_w, wp_o, 128, 128, 1, 16384);

    // 2) residual = bn(conv1x1 stride2)
    sc_kernel<<<16384 * 128 / 256, B256, 0, stream>>>(x, sc_w, sc_b, sc_g, sc_bt, resid);

    // 3) ic1: 1x1 conv 64->128 on 64x64 (WMMA), bn+relu -> h1 fp32
    conv_gemm<<<dim3(2048, 4, 1), B32, 0, stream>>>(Ax, wp_ic1, h1, nullptr, 0,
        ic1_b, ic1_g, ic1_bt, 1, 64, 128, 12, 6, 1, 1, 0);

    // 4) ic2: depthwise 3x3 stride2, bn+relu -> h fp32 + hb f16
    dwconv_kernel<<<2097152 / 256, B256, 0, stream>>>(h1, 64, 64, h, hb, nullptr, nullptr,
        32, 32, ic2_w, ic2_b, ic2_g, ic2_bt, 1, 0, 128, 3, 1, 2, 2097152);

    // 5) multi-scale depthwise branch -> xmulti
    dwconv_kernel<<<(704512 + 255) / 256, B256, 0, stream>>>(h, 32, 32, xmulti, nullptr, nullptr,
        nullptr, 32, 32, dt3_w, dt3_b, dt3_g, dt3_bt, 1, 0, 43, 3, 1, 1, 704512);
    dwconv_kernel<<<(704512 + 255) / 256, B256, 0, stream>>>(h, 32, 32, xmulti, nullptr, nullptr,
        nullptr, 32, 32, dt5_w, dt5_b, dt5_g, dt5_bt, 1, 43, 43, 5, 2, 1, 704512);
    dwconv_kernel<<<(688128 + 255) / 256, B256, 0, stream>>>(h, 32, 32, xmulti, nullptr, nullptr,
        nullptr, 32, 32, dt7_w, dt7_b, dt7_g, dt7_bt, 1, 86, 42, 7, 3, 1, 688128);

    // 6) gating
    gfeat_kernel<<<16, 128, 0, stream>>>(xmulti, gfeat);
    gate_kernel<<<1, 32, 0, stream>>>(gfeat, fc_w, fc_b, gmask);

    // 7) md1/md3/md5 implicit-GEMM convs (WMMA) -> fm f16 [16384][384]
    conv_gemm<<<dim3(512, 4, 1), B32, 0, stream>>>(hb, wp_md1, nullptr, fm + 0, 384,
        md1_b, nullptr, nullptr, 0, 128, 128, 10, 5, 1, 1, 0);
    conv_gemm<<<dim3(512, 4, 1), B32, 0, stream>>>(hb, wp_md3, nullptr, fm + 128, 384,
        md3_b, nullptr, nullptr, 0, 128, 128, 10, 5, 3, 3, 1);
    conv_gemm<<<dim3(512, 4, 1), B32, 0, stream>>>(hb, wp_md5, nullptr, fm + 256, 384,
        md5_b, nullptr, nullptr, 0, 128, 128, 10, 5, 5, 5, 2);

    // 8) Q/K/V projections (WMMA)
    gemm_nt<<<dim3(512, 2, 1), B32, 0, stream>>>(fm, 384, 0, wp_q, 384, 0,
        nullptr, 0, 0, Qb, 64, 0, q_b, 0, 384);
    gemm_nt<<<dim3(512, 2, 1), B32, 0, stream>>>(fm, 384, 0, wp_k, 384, 0,
        nullptr, 0, 0, Kb, 64, 0, k_b, 0, 384);
    gemm_nt<<<dim3(512, 4, 1), B32, 0, stream>>>(fm, 384, 0, wp_v, 384, 0,
        nullptr, 0, 0, Vb, 128, 0, v_b, 0, 384);

    // 9) scores = Q K^T per batch (WMMA, batched over z)
    gemm_nt<<<dim3(32, 32, 16), B32, 0, stream>>>(Qb, 64, 65536, Kb, 64, 65536,
        scores, 1024, 1048576, nullptr, 0, 0, nullptr, 0, 64);

    // 10) softmax, key importance, top-2 keys, rank-2 attn@V
    softmax_rows<<<16384, B256, 0, stream>>>(scores, 1024, ATTN_SCALE);
    colsum_kernel<<<dim3(16, 4, 1), B256, 0, stream>>>(scores, keyimp);
    top2_kernel<<<1, 32, 0, stream>>>(keyimp, kidx);
    sparse_av<<<2097152 / 256, B256, 0, stream>>>(scores, Vb, kidx, attnout);

    // 11) output projection (WMMA) -> out_md fp32
    gemm_nt<<<dim3(512, 4, 1), B32, 0, stream>>>(attnout, 128, 0, wp_o, 128, 0,
        out_md, 128, 0, nullptr, 0, 0, o_b, 0, 128);

    // 12) y = out_dt * out_md  (gate applied per channel-group) -> f16
    ymul_kernel<<<2097152 / 256, B256, 0, stream>>>(xmulti, out_md, gmask, ymul);

    // 13) oc1: 1x1 conv (WMMA) + bn + relu -> y1 fp32
    conv_gemm<<<dim3(512, 4, 1), B32, 0, stream>>>(ymul, wp_oc1, y1, nullptr, 0,
        oc1_b, oc1_g, oc1_bt, 1, 128, 128, 10, 5, 1, 1, 0);

    // 14) oc2: depthwise 3x3 + bn + relu + residual, write NCHW fp32 output
    dwconv_kernel<<<2097152 / 256, B256, 0, stream>>>(y1, 32, 32, nullptr, nullptr,
        (float*)d_out, resid, 32, 32, oc2_w, oc2_b, oc2_g, oc2_bt, 1, 0, 128, 3, 1, 1, 2097152);
}
